// DualCopyGenerator_49692771614990
// MI455X (gfx1250) — matrix operand order repaired and checked by reading, not compiled, not run on hardware
//
#include <hip/hip_runtime.h>
#include <hip/hip_bf16.h>
#include <math.h>

#define BB   8
#define TT   64
#define SS   128
#define TDIM 512
#define SDIM 512
#define VV   32000
#define EXTD 32256          // V + S1 + S2
#define ROWS (BB * TT)      // 512
#define LNEPS 1e-5f

typedef __attribute__((ext_vector_type(16))) _Float16 v16h;
typedef __attribute__((ext_vector_type(8)))  float    v8f;

union HFrag { v16h v; float4 f[2]; };

// ---------------------------------------------------------------------------
// Generic WMMA GEMM: C[m,n] = sum_k A[m,k]*B[n,k] (+bias[n]); A:MxK, B:NxK f16
// Block = 256 threads = 8 waves; wave tile 32x64; block tile 64x256.
// transOut: write C to out[((m>>7)*N + n)*128 + (m&127)]  (i.e. (B,E,S) layout)
// ---------------------------------------------------------------------------
__global__ __launch_bounds__(256)
void wmma_gemm_nt(const _Float16* __restrict__ A, const _Float16* __restrict__ Bm,
                  float* __restrict__ C, const float* __restrict__ bias,
                  int M, int N, int K, int transOut)
{
    const int lane = threadIdx.x & 31;
    const int wave = threadIdx.x >> 5;
    const int wm   = wave & 1;                    // 2 waves along M
    const int wn   = wave >> 1;                   // 4 waves along N
    const int m0   = blockIdx.x * 64  + wm * 32;  // wave M base (2 x 16)
    const int n0   = blockIdx.y * 256 + wn * 64;  // wave N base (4 x 16)

    v8f zero = {0.f,0.f,0.f,0.f,0.f,0.f,0.f,0.f};
    v8f acc[2][4];
    #pragma unroll
    for (int i = 0; i < 2; ++i)
        #pragma unroll
        for (int j = 0; j < 4; ++j) acc[i][j] = zero;

    // A fragment addressing (16x32 f16, ISA 7.12.2): lane<16 -> M=lane, K={0..7,16..23}
    //                                                lane>=16 -> M=lane-16, K={8..15,24..31}
    const int arow  = m0 + (lane & 15);
    const int akoff = (lane < 16) ? 0 : 8;
    // B fragment (32x16, K-major per row n): lane<16 -> K={0..7,16..23} ; lane>=16 -> K={8..15,24..31}
    const int bcol  = lane & 15;
    const int bkoff = (lane < 16) ? 0 : 8;

    for (int k0 = 0; k0 < K; k0 += 32) {
        HFrag a[2];
        #pragma unroll
        for (int i = 0; i < 2; ++i) {
            const _Float16* ap = A + (size_t)(arow + 16 * i) * K + k0 + akoff;
            a[i].f[0] = *(const float4*)(ap);        // K +0..7
            a[i].f[1] = *(const float4*)(ap + 16);   // K +16..23
        }
        HFrag b[4];
        #pragma unroll
        for (int j = 0; j < 4; ++j) {
            const _Float16* bp = Bm + (size_t)(n0 + 16 * j + bcol) * K + k0 + bkoff;
            b[j].f[0] = ((const float4*)bp)[0];      // K +0..7
            b[j].f[1] = ((const float4*)bp)[2];      // K +16..23
        }
        #pragma unroll
        for (int i = 0; i < 2; ++i)
            #pragma unroll
            for (int j = 0; j < 4; ++j)
                acc[i][j] = __builtin_amdgcn_wmma_f32_16x16x32_f16(
                    false, a[i].v, false, b[j].v, (short)0, acc[i][j], false, false);
    }

    // C layout (ISA 7.12.2): VGPR r holds M = r (lanes 0-15) / M = 8+r (lanes 16-31), N = lane&15
    const int crow = (lane < 16) ? 0 : 8;
    const int ccol = lane & 15;
    #pragma unroll
    for (int i = 0; i < 2; ++i) {
        #pragma unroll
        for (int j = 0; j < 4; ++j) {
            const int n  = n0 + 16 * j + ccol;
            const float bv = bias ? bias[n] : 0.0f;
            #pragma unroll
            for (int r = 0; r < 8; ++r) {
                const int m = m0 + 16 * i + crow + r;
                const float val = acc[i][j][r] + bv;
                if (!transOut) C[(size_t)m * N + n] = val;
                else           C[((size_t)(m >> 7) * N + n) * 128 + (m & 127)] = val;
            }
        }
    }
}

// ---------------------------------------------------------------------------
__global__ void cvt_f32_f16(const float* __restrict__ s, _Float16* __restrict__ d, int n)
{
    int i = blockIdx.x * 256 + threadIdx.x;
    if (i < n) d[i] = (_Float16)s[i];
}

// attn_w (TD, SD+TD) -> packed Wq (TD,TD), Wk (TD,SD) in f16
__global__ void split_attnw(const float* __restrict__ aw,
                            _Float16* __restrict__ wq, _Float16* __restrict__ wk)
{
    int i = blockIdx.x * 256 + threadIdx.x;
    if (i < TDIM * TDIM) {
        int e = i >> 9, d = i & 511;
        wq[i] = (_Float16)aw[(size_t)e * 1024 + d];
        wk[i] = (_Float16)aw[(size_t)e * 1024 + 512 + d];
    }
}

// masks: rows 0..511 tgt, 512..1535 src1, 1536..2559 src2 (all len-512 rows)
__global__ __launch_bounds__(256)
void mask_kernel(const float* __restrict__ tgt, const float* __restrict__ s1,
                 const float* __restrict__ s2, float* mt, float* m1, float* m2)
{
    __shared__ float red[256];
    int row = blockIdx.x;
    const float* src; float* dst;
    if (row < 512)       { src = tgt + (size_t)row * 512;          dst = mt + row; }
    else if (row < 1536) { int r = row - 512;  src = s1 + (size_t)r * 512; dst = m1 + r; }
    else                 { int r = row - 1536; src = s2 + (size_t)r * 512; dst = m2 + r; }
    float a = 0.f;
    for (int j = threadIdx.x; j < 512; j += 256) a += fabsf(src[j]);
    red[threadIdx.x] = a; __syncthreads();
    for (int o = 128; o > 0; o >>= 1) {
        if (threadIdx.x < o) red[threadIdx.x] += red[threadIdx.x + o];
        __syncthreads();
    }
    if (threadIdx.x == 0) *dst = (red[0] > 0.f) ? 1.f : 0.f;
}

// one block per (b,t); 128 threads = S. Computes tanh-scores, LN(att), softmax.
__global__ __launch_bounds__(128)
void score_kernel(const float* __restrict__ sq,   // (512,512)
                  const float* __restrict__ skT,  // (8,512,128)
                  const float* __restrict__ bias, // (512)
                  const float* __restrict__ vvec, // (512)
                  const float* __restrict__ kmask,// (8*128)
                  const float* __restrict__ qmask,// (512)
                  float* __restrict__ att_ln,     // (512,128)
                  float* __restrict__ attnp)      // (512,128)
{
    __shared__ float sqs[512], vb[512], bb[512], red[128];
    const int bt = blockIdx.x, b = bt >> 6, s = threadIdx.x;
    for (int j = s; j < 512; j += 128) {
        sqs[j] = sq[(size_t)bt * 512 + j];
        vb[j]  = vvec[j];
        bb[j]  = bias[j];
    }
    __syncthreads();
    const float* skp = skT + (size_t)b * 512 * 128 + s;  // coalesced across s
    float acc = 0.f;
    for (int e = 0; e < 512; ++e)
        acc += vb[e] * tanhf(sqs[e] + bb[e] + skp[(size_t)e * 128]);
    const float km = kmask[b * 128 + s], qm = qmask[bt];
    const float oa = acc * km * qm;

    // LayerNorm over S
    red[s] = oa; __syncthreads();
    for (int o = 64; o > 0; o >>= 1) { if (s < o) red[s] += red[s + o]; __syncthreads(); }
    const float mean = red[0] * (1.f / 128.f); __syncthreads();
    const float dv = oa - mean;
    red[s] = dv * dv; __syncthreads();
    for (int o = 64; o > 0; o >>= 1) { if (s < o) red[s] += red[s + o]; __syncthreads(); }
    const float var = red[0] * (1.f / 128.f); __syncthreads();
    att_ln[(size_t)bt * 128 + s] = dv * rsqrtf(var + LNEPS);

    // masked softmax over S (uses raw scores)
    const float sc = (km == 0.f) ? -__builtin_inff() : acc;
    red[s] = sc; __syncthreads();
    for (int o = 64; o > 0; o >>= 1) { if (s < o) red[s] = fmaxf(red[s], red[s + o]); __syncthreads(); }
    const float mx = red[0]; __syncthreads();
    const float ex = expf(sc - mx);
    red[s] = ex; __syncthreads();
    for (int o = 64; o > 0; o >>= 1) { if (s < o) red[s] += red[s + o]; __syncthreads(); }
    attnp[(size_t)bt * 128 + s] = ex / red[0];
}

// c[b,t,d] = qmask * sum_s attn[b,t,s] * k[b,s,d]
__global__ __launch_bounds__(256)
void weight_kernel(const float* __restrict__ attnp, const float* __restrict__ kf,
                   const float* __restrict__ qmask, float* __restrict__ cout)
{
    __shared__ float ap[128];
    const int bt = blockIdx.x, b = bt >> 6, tid = threadIdx.x;
    if (tid < 128) ap[tid] = attnp[(size_t)bt * 128 + tid];
    __syncthreads();
    const float qm = qmask[bt];
    for (int d = tid; d < 512; d += 256) {
        float acc = 0.f;
        const float* kp = kf + (size_t)b * 128 * 512 + d;
        for (int s2 = 0; s2 < 128; ++s2) acc += ap[s2] * kp[(size_t)s2 * 512];
        cout[(size_t)bt * 512 + d] = acc * qm;
    }
}

// gate probs p[bt,3] = softmax(concat(tgt,c1,c2) @ gate_w^T + gate_b)
__global__ __launch_bounds__(128)
void gate_kernel(const float* __restrict__ tgt, const float* __restrict__ c1,
                 const float* __restrict__ c2, const float* __restrict__ gw,
                 const float* __restrict__ gb, float* __restrict__ p)
{
    __shared__ float red[3][128];
    const int bt = blockIdx.x, tid = threadIdx.x;
    float g0 = 0.f, g1 = 0.f, g2 = 0.f;
    for (int d = tid; d < 1536; d += 128) {
        float x = (d < 512)  ? tgt[(size_t)bt * 512 + d]
                : (d < 1024) ? c1[(size_t)bt * 512 + d - 512]
                             : c2[(size_t)bt * 512 + d - 1024];
        g0 += gw[d] * x; g1 += gw[1536 + d] * x; g2 += gw[3072 + d] * x;
    }
    red[0][tid] = g0; red[1][tid] = g1; red[2][tid] = g2; __syncthreads();
    for (int o = 64; o > 0; o >>= 1) {
        if (tid < o) { red[0][tid] += red[0][tid+o]; red[1][tid] += red[1][tid+o]; red[2][tid] += red[2][tid+o]; }
        __syncthreads();
    }
    if (tid == 0) {
        float a = red[0][0] + gb[0], b2 = red[1][0] + gb[1], c = red[2][0] + gb[2];
        float m = fmaxf(a, fmaxf(b2, c));
        float ea = expf(a - m), eb = expf(b2 - m), ec = expf(c - m);
        float s = ea + eb + ec;
        p[bt*3+0] = ea/s; p[bt*3+1] = eb/s; p[bt*3+2] = ec/s;
    }
}

// per-row mean / rstd over V=32000 logits
__global__ __launch_bounds__(256)
void rowstats_kernel(const float* __restrict__ logits, float* mean, float* rstd)
{
    __shared__ float rs[256], rq[256];
    const int row = blockIdx.x, tid = threadIdx.x;
    const float* p = logits + (size_t)row * VV;
    float s = 0.f, q = 0.f;
    for (int j = tid; j < VV; j += 256) { float x = p[j]; s += x; q += x * x; }
    rs[tid] = s; rq[tid] = q; __syncthreads();
    for (int o = 128; o > 0; o >>= 1) {
        if (tid < o) { rs[tid] += rs[tid+o]; rq[tid] += rq[tid+o]; }
        __syncthreads();
    }
    if (tid == 0) {
        float m = rs[0] / (float)VV;
        float v = rq[0] / (float)VV - m * m;
        mean[row] = m; rstd[row] = rsqrtf(v + LNEPS);
    }
}

// out[row,j] = p0 * LN(logits)[row,j]  (j<V),  0 for pad region
__global__ __launch_bounds__(256)
void combine_kernel(const float* __restrict__ logits, const float* __restrict__ mean,
                    const float* __restrict__ rstd, const float* __restrict__ p,
                    float* __restrict__ out)
{
    const size_t idx = (size_t)blockIdx.x * 256 + threadIdx.x;
    const int row = (int)(idx / EXTD), j = (int)(idx % EXTD);
    float val = 0.f;
    if (j < VV)
        val = p[row * 3] * (logits[(size_t)row * VV + j] - mean[row]) * rstd[row];
    out[idx] = val;
}

// out[bt, map[b,s]] += p[bt, 1+src] * att[bt, s]
__global__ __launch_bounds__(256)
void scatter_kernel(const float* __restrict__ att1, const float* __restrict__ att2,
                    const int* __restrict__ map1, const int* __restrict__ map2,
                    const float* __restrict__ p, float* __restrict__ out)
{
    const int idx = blockIdx.x * 256 + threadIdx.x;   // 0 .. 2*B*T*S-1
    const int which = idx >> 16;                      // 65536 items per source
    const int i = idx & 0xFFFF;
    const int b = i >> 13;                            // T*S = 8192 per b
    const int ts = i & 8191;
    const int t = ts >> 7, s = ts & 127;
    const int bt = b * TT + t;
    const float* att = which ? att2 : att1;
    const int* mp = which ? map2 : map1;
    const float val = p[bt * 3 + 1 + which] * att[(size_t)bt * 128 + s];
    const int col = mp[b * 128 + s];
    atomicAdd(&out[(size_t)bt * EXTD + col], val);
}

// ---------------------------------------------------------------------------
extern "C" void kernel_launch(void* const* d_in, const int* in_sizes, int n_in,
                              void* d_out, int out_size, void* d_ws, size_t ws_size,
                              hipStream_t stream)
{
    (void)in_sizes; (void)n_in; (void)out_size; (void)ws_size;
    const float* tgt   = (const float*)d_in[0];   // (512,512)
    const float* src1  = (const float*)d_in[1];   // (8,128,512)
    const float* src2  = (const float*)d_in[2];
    const int*   map1  = (const int*)d_in[3];     // (8,128)
    const int*   map2  = (const int*)d_in[4];
    const float* Wout  = (const float*)d_in[5];   // (32000,512)
    const float* bout  = (const float*)d_in[6];   // (32000)
    const float* aw1   = (const float*)d_in[7];   // (512,1024)
    const float* ab1   = (const float*)d_in[8];
    const float* v1    = (const float*)d_in[9];
    const float* aw2   = (const float*)d_in[10];
    const float* ab2   = (const float*)d_in[11];
    const float* v2    = (const float*)d_in[12];
    const float* gw    = (const float*)d_in[13];  // (3,1536)
    const float* gb    = (const float*)d_in[14];
    float* out = (float*)d_out;

    // workspace carve
    size_t off = 0;
    char* base = (char*)d_ws;
    auto alloc = [&](size_t bytes) -> void* {
        void* p = base + off; off += (bytes + 255) & ~(size_t)255; return p;
    };
    _Float16* tgt_h  = (_Float16*)alloc((size_t)ROWS * TDIM * 2);
    _Float16* W_h    = (_Float16*)alloc((size_t)VV * TDIM * 2);
    _Float16* s1_h   = (_Float16*)alloc((size_t)BB * SS * SDIM * 2);
    _Float16* s2_h   = (_Float16*)alloc((size_t)BB * SS * SDIM * 2);
    _Float16* wq1_h  = (_Float16*)alloc((size_t)TDIM * TDIM * 2);
    _Float16* wk1_h  = (_Float16*)alloc((size_t)TDIM * SDIM * 2);
    _Float16* wq2_h  = (_Float16*)alloc((size_t)TDIM * TDIM * 2);
    _Float16* wk2_h  = (_Float16*)alloc((size_t)TDIM * SDIM * 2);
    float* logits = (float*)alloc((size_t)ROWS * VV * 4);
    float* sq1    = (float*)alloc((size_t)ROWS * TDIM * 4);
    float* sq2    = (float*)alloc((size_t)ROWS * TDIM * 4);
    float* skT1   = (float*)alloc((size_t)BB * TDIM * SS * 4);
    float* skT2   = (float*)alloc((size_t)BB * TDIM * SS * 4);
    float* att1   = (float*)alloc((size_t)ROWS * SS * 4);
    float* att2   = (float*)alloc((size_t)ROWS * SS * 4);
    float* ap1    = (float*)alloc((size_t)ROWS * SS * 4);
    float* ap2    = (float*)alloc((size_t)ROWS * SS * 4);
    float* c1     = (float*)alloc((size_t)ROWS * TDIM * 4);
    float* c2     = (float*)alloc((size_t)ROWS * TDIM * 4);
    float* pg     = (float*)alloc((size_t)ROWS * 3 * 4);
    float* mtgt   = (float*)alloc((size_t)ROWS * 4);
    float* m1     = (float*)alloc((size_t)BB * SS * 4);
    float* m2     = (float*)alloc((size_t)BB * SS * 4);
    float* rmean  = (float*)alloc((size_t)ROWS * 4);
    float* rrstd  = (float*)alloc((size_t)ROWS * 4);

    // 1) conversions
    {
        int n = ROWS * TDIM;
        cvt_f32_f16<<<(n + 255) / 256, 256, 0, stream>>>(tgt, tgt_h, n);
        n = VV * TDIM;
        cvt_f32_f16<<<(n + 255) / 256, 256, 0, stream>>>(Wout, W_h, n);
        n = BB * SS * SDIM;
        cvt_f32_f16<<<(n + 255) / 256, 256, 0, stream>>>(src1, s1_h, n);
        cvt_f32_f16<<<(n + 255) / 256, 256, 0, stream>>>(src2, s2_h, n);
        n = TDIM * TDIM;
        split_attnw<<<(n + 255) / 256, 256, 0, stream>>>(aw1, wq1_h, wk1_h);
        split_attnw<<<(n + 255) / 256, 256, 0, stream>>>(aw2, wq2_h, wk2_h);
    }

    // 2) masks (512 + 1024 + 1024 rows)
    mask_kernel<<<2560, 256, 0, stream>>>(tgt, src1, src2, mtgt, m1, m2);

    // 3) WMMA GEMMs
    wmma_gemm_nt<<<dim3(ROWS / 64, VV / 256), 256, 0, stream>>>(
        tgt_h, W_h, logits, bout, ROWS, VV, TDIM, 0);
    wmma_gemm_nt<<<dim3(ROWS / 64, TDIM / 256), 256, 0, stream>>>(
        tgt_h, wq1_h, sq1, nullptr, ROWS, TDIM, TDIM, 0);
    wmma_gemm_nt<<<dim3(ROWS / 64, TDIM / 256), 256, 0, stream>>>(
        tgt_h, wq2_h, sq2, nullptr, ROWS, TDIM, TDIM, 0);
    wmma_gemm_nt<<<dim3((BB * SS) / 64, TDIM / 256), 256, 0, stream>>>(
        s1_h, wk1_h, skT1, nullptr, BB * SS, TDIM, SDIM, 1);
    wmma_gemm_nt<<<dim3((BB * SS) / 64, TDIM / 256), 256, 0, stream>>>(
        s2_h, wk2_h, skT2, nullptr, BB * SS, TDIM, SDIM, 1);

    // 4) tanh-attention scores + LN + masked softmax
    score_kernel<<<ROWS, 128, 0, stream>>>(sq1, skT1, ab1, v1, m1, mtgt, att1, ap1);
    score_kernel<<<ROWS, 128, 0, stream>>>(sq2, skT2, ab2, v2, m2, mtgt, att2, ap2);

    // 5) context vectors
    weight_kernel<<<ROWS, 256, 0, stream>>>(ap1, src1, mtgt, c1);
    weight_kernel<<<ROWS, 256, 0, stream>>>(ap2, src2, mtgt, c2);

    // 6) gate probs
    gate_kernel<<<ROWS, 128, 0, stream>>>(tgt, c1, c2, gw, gb, pg);

    // 7) logits LayerNorm stats, gated combine, scatter-add
    rowstats_kernel<<<ROWS, 256, 0, stream>>>(logits, rmean, rrstd);
    combine_kernel<<<(ROWS * EXTD) / 256, 256, 0, stream>>>(logits, rmean, rrstd, pg, out);
    scatter_kernel<<<(2 * ROWS * SS) / 256, 256, 0, stream>>>(att1, att2, map1, map2, pg, out);
}
